// MovingAverageGatedAttention_52604759442088
// MI455X (gfx1250) — matrix operand order repaired and checked by reading, not compiled, not run on hardware
//
#include <hip/hip_runtime.h>
#include <hip/hip_bf16.h>

// ---------------------------------------------------------------------------
// MEGA block forward for MI455X (gfx1250, wave32, WMMA).
// All GEMMs use v_wmma_f32_16x16x32_f16 with fragment layouts per CDNA5 ISA
// 7.12.2. Sequential scans (timestep-norm cumsum, 16-state EMA) are wave32
// scan kernels. Attention uses two-pass online softmax + materialized probs.
// ---------------------------------------------------------------------------

#define Bx 8
#define Lx 2048
#define Dx 1024
#define Zx 128
#define Hx 2048
#define NEMA 16
#define Gx 32
#define MG 32          // Dx / Gx
#define OC 4224        // Z + H + 2D
#define MAXPOSx 2048
#define EPSx 1e-5f
#define PRIORC 2.0f

typedef _Float16 f16;
typedef _Float16 v16h __attribute__((ext_vector_type(16)));
typedef float    v8f  __attribute__((ext_vector_type(8)));

union FragAB { uint4 u[2]; v16h v; };

__device__ __forceinline__ float sigm_(float x) { return 1.0f / (1.0f + __expf(-x)); }
__device__ __forceinline__ float silu_(float x) { return x / (1.0f + __expf(-x)); }
__device__ __forceinline__ v8f zero8() {
  v8f z = {0.f,0.f,0.f,0.f,0.f,0.f,0.f,0.f};
  return z;
}

// A fragment: 16 rows (row0..+15) x 32 K (k0..+31) from row-major f16, ld elems.
// Lanes 0-15: M=lane, K chunks {k0..k0+7, k0+16..k0+23};
// lanes 16-31: M=lane-16, K chunks {k0+8.., k0+24..}.   (ISA 7.12.2)
__device__ __forceinline__ v16h load_a(const f16* A, long row0, long k0, long ld) {
  int lane = threadIdx.x & 31;
  int m    = lane & 15;
  int hi   = lane >> 4;
  const f16* p = A + (row0 + m) * ld + k0 + hi * 8;
  FragAB f;
  f.u[0] = *(const uint4*)(p);
  f.u[1] = *(const uint4*)(p + 16);
  return f.v;
}

// B fragment B[k][n] for tile (k0, col0), sourced from B' stored (N,K) row-major
// (i.e. B'[n][k] = B[k][n]).  Lanes 0-15: N=lane, K chunks {k0..k0+7, k0+16..k0+23};
// lanes 16-31: N=lane-16, K chunks {k0+8.., k0+24..}.
__device__ __forceinline__ v16h load_b(const f16* Bp, long col0, long k0, long ld) {
  int lane = threadIdx.x & 31;
  int n    = lane & 15;
  int hi   = lane >> 4;
  const f16* p = Bp + (col0 + n) * ld + k0 + hi * 8;
  FragAB f;
  f.u[0] = *(const uint4*)(p);
  f.u[1] = *(const uint4*)(p + 16);
  return f.v;
}

__device__ __forceinline__ v8f wmma_(v16h a, v16h b, v8f c) {
  return __builtin_amdgcn_wmma_f32_16x16x32_f16(false, a, false, b, (short)0, c, false, false);
}

// ---------------------------------------------------------------------------
// Elementwise helpers
// ---------------------------------------------------------------------------
__global__ void cvt_f32_f16_kernel(const float* __restrict__ src, f16* __restrict__ dst, long n) {
  long i = (long)blockIdx.x * 256 + threadIdx.x;
  if (i < n) dst[i] = (f16)src[i];
}

__global__ void zero16_kernel(uint4* __restrict__ p, long n16) {
  long i = (long)blockIdx.x * 256 + threadIdx.x;
  if (i < n16) {
    uint4 z; z.x = 0u; z.y = 0u; z.z = 0u; z.w = 0u;
    p[i] = z;
  }
}

// ---------------------------------------------------------------------------
// TimestepNorm: causal cumulative group norm. One wave32 per (b, group);
// lane j owns channel d = g*32 + j; group sums via wave32 butterfly.
// ---------------------------------------------------------------------------
__global__ void __launch_bounds__(32)
tsnorm_kernel(const float* __restrict__ x, const float* __restrict__ pm,
              const float* __restrict__ plv, const float* __restrict__ w,
              const float* __restrict__ bias, f16* __restrict__ xn16) {
  int bg = blockIdx.x;
  int b = bg / Gx, g = bg % Gx;
  int lane = threadIdx.x;
  int d = g * MG + lane;
  float wt = w[d], bs = bias[d];
  float pmean = pm[g];
  float pv = __expf(plv[g]);
  float s1 = 0.f, s2 = 0.f;
  const float* xp = x + ((long)b * Lx) * Dx + d;
  f16* op = xn16 + ((long)b * Lx) * Dx + d;
  for (int t = 0; t < Lx; ++t) {
    float v = xp[(long)t * Dx];
    float gs = v, gs2 = v * v;
    #pragma unroll
    for (int m = 1; m < 32; m <<= 1) { gs += __shfl_xor(gs, m); gs2 += __shfl_xor(gs2, m); }
    s1 += gs; s2 += gs2;
    float cnt  = PRIORC + (float)MG * (float)(t + 1);
    float mean = (PRIORC * pmean + s1) / cnt;
    float var  = (PRIORC * (pv + pmean * pmean) + s2) / cnt - mean * mean;
    float xnv  = (v - mean) * rsqrtf(var + EPSx);
    op[(long)t * Dx] = (f16)(xnv * wt + bs);
  }
}

// ---------------------------------------------------------------------------
// MultiHeadEMA as a 16-state linear recurrence (equivalent to the FFT conv):
//   s_n[t] = q_n s_n[t-1] + (p*beta)_n x[t];  y[t] = omega x[t] + sum_n wo_n s_n[t]
// One lane per (b,d).
// ---------------------------------------------------------------------------
__global__ void __launch_bounds__(256)
ema_kernel(const f16* __restrict__ xn16, const float* __restrict__ delta,
           const float* __restrict__ alpha, const float* __restrict__ beta,
           const float* __restrict__ gamma, const float* __restrict__ omega,
           f16* __restrict__ mx16) {
  int idx = blockIdx.x * 256 + threadIdx.x;           // 0..B*D-1
  int b = idx / Dx, d = idx % Dx;
  float qn[NEMA], wi[NEMA], wo[NEMA], st[NEMA];
  #pragma unroll
  for (int n = 0; n < NEMA; ++n) {
    float p = sigm_(delta[d * NEMA + n]);
    float a = sigm_(alpha[d * NEMA + n]);
    qn[n] = 1.0f - p * a;
    wi[n] = p * beta[d * NEMA + n];
    wo[n] = gamma[d * NEMA + n] * 0.25f;              // 1/sqrt(N=16)
    st[n] = 0.f;
  }
  float om = omega[d];
  const f16* xp = xn16 + ((long)b * Lx) * Dx + d;
  f16* op = mx16 + ((long)b * Lx) * Dx + d;
  for (int t = 0; t < Lx; ++t) {
    float xv = (float)xp[(long)t * Dx];
    float y = om * xv;
    #pragma unroll
    for (int n = 0; n < NEMA; ++n) { st[n] = qn[n] * st[n] + wi[n] * xv; y += wo[n] * st[n]; }
    op[(long)t * Dx] = (f16)y;
  }
}

// RMS-norm in place over D. One wave32 per (b,l) row.
__global__ void __launch_bounds__(128)
rms_kernel(f16* __restrict__ mx16, const float* __restrict__ rw) {
  int wave = threadIdx.x >> 5, lane = threadIdx.x & 31;
  long row = (long)blockIdx.x * 4 + wave;
  f16* p = mx16 + row * Dx;
  float ss = 0.f;
  for (int i = lane; i < Dx; i += 32) { float v = (float)p[i]; ss += v * v; }
  #pragma unroll
  for (int m = 1; m < 32; m <<= 1) ss += __shfl_xor(ss, m);
  float scale = rsqrtf(ss * (1.0f / Dx) + EPSx);
  for (int i = lane; i < Dx; i += 32) p[i] = (f16)((float)p[i] * scale * rw[i]);
}

// ---------------------------------------------------------------------------
// GEMM stage 1: v = silu(xn @ Wv^T + bv), written transposed to vT (B,H,L) f16.
// 128 threads = 4 waves; each wave computes a 32x32 C tile (block = 64x64).
// ---------------------------------------------------------------------------
__global__ void __launch_bounds__(128)
gemm_v_kernel(const f16* __restrict__ A, const f16* __restrict__ W,
              const float* __restrict__ bv, f16* __restrict__ vT) {
  int wave = threadIdx.x >> 5;
  long m0 = (long)blockIdx.x * 64 + (wave >> 1) * 32;   // token rows in [0, B*L)
  long n0 = (long)blockIdx.y * 64 + (wave & 1) * 32;    // H cols
  v8f c00 = zero8(), c01 = zero8(), c10 = zero8(), c11 = zero8();
  for (int k = 0; k < Dx; k += 32) {
    if (k + 32 < Dx) { __builtin_prefetch(A + m0 * Dx + k + 32, 0, 1); }
    v16h a0 = load_a(A, m0,      k, Dx);
    v16h a1 = load_a(A, m0 + 16, k, Dx);
    v16h b0 = load_b(W, n0,      k, Dx);
    v16h b1 = load_b(W, n0 + 16, k, Dx);
    c00 = wmma_(a0, b0, c00); c01 = wmma_(a0, b1, c01);
    c10 = wmma_(a1, b0, c10); c11 = wmma_(a1, b1, c11);
  }
  int lane = threadIdx.x & 31, nlo = lane & 15, hi = lane >> 4;
  v8f cs[4] = {c00, c01, c10, c11};
  int moff[4] = {0, 0, 16, 16}, noff[4] = {0, 16, 0, 16};
  #pragma unroll
  for (int t = 0; t < 4; ++t) {
    #pragma unroll
    for (int i = 0; i < 8; ++i) {
      long row = m0 + moff[t] + i + 8 * hi;
      long col = n0 + noff[t] + nlo;
      float val = silu_(cs[t][i] + bv[col]);
      long b = row >> 11, l = row & (Lx - 1);
      vT[((long)b * Hx + col) * Lx + l] = (f16)val;
    }
  }
}

// ---------------------------------------------------------------------------
// GEMM stage 2: base = mx @ Wmx^T + bmx with fused gating epilogue:
//   cols [0,1024)      -> u  = sigmoid          (f16)
//   cols [1024,1152)   -> z  = silu, then q/k affine+scale   (f16)
//   cols [1152,3200)   -> r  = silu             (f16)
//   cols [3200,4224)   -> hx raw                (f16)
// ---------------------------------------------------------------------------
__global__ void __launch_bounds__(128)
gemm_base_kernel(const f16* __restrict__ A, const f16* __restrict__ W,
                 const float* __restrict__ bmx, const float* __restrict__ qkg,
                 const float* __restrict__ qkb, f16* __restrict__ u16,
                 f16* __restrict__ q16, f16* __restrict__ k16,
                 f16* __restrict__ r16, f16* __restrict__ hx16) {
  int wave = threadIdx.x >> 5;
  long m0 = (long)blockIdx.x * 64 + (wave >> 1) * 32;
  long n0 = (long)blockIdx.y * 64 + (wave & 1) * 32;
  v8f c00 = zero8(), c01 = zero8(), c10 = zero8(), c11 = zero8();
  for (int k = 0; k < Dx; k += 32) {
    v16h a0 = load_a(A, m0,      k, Dx);
    v16h a1 = load_a(A, m0 + 16, k, Dx);
    v16h b0 = load_b(W, n0,      k, Dx);
    v16h b1 = load_b(W, n0 + 16, k, Dx);
    c00 = wmma_(a0, b0, c00); c01 = wmma_(a0, b1, c01);
    c10 = wmma_(a1, b0, c10); c11 = wmma_(a1, b1, c11);
  }
  int lane = threadIdx.x & 31, nlo = lane & 15, hi = lane >> 4;
  v8f cs[4] = {c00, c01, c10, c11};
  int moff[4] = {0, 0, 16, 16}, noff[4] = {0, 16, 0, 16};
  #pragma unroll
  for (int t = 0; t < 4; ++t) {
    for (int i = 0; i < 8; ++i) {
      long bl  = m0 + moff[t] + i + 8 * hi;           // b*L + l
      long col = n0 + noff[t] + nlo;
      float val = cs[t][i] + bmx[col];
      if (col < Dx) {
        u16[bl * Dx + col] = (f16)sigm_(val);
      } else if (col < Dx + Zx) {
        int zi = (int)(col - Dx);
        float z = silu_(val);
        q16[bl * Zx + zi] = (f16)((z * qkg[zi]      + qkb[zi])      * 0.08838834764831845f);
        k16[bl * Zx + zi] = (f16)( z * qkg[Zx + zi] + qkb[Zx + zi]);
      } else if (col < Dx + Zx + Hx) {
        r16[bl * Hx + (col - Dx - Zx)] = (f16)silu_(val);
      } else {
        hx16[bl * Dx + (col - Dx - Zx - Hx)] = (f16)val;
      }
    }
  }
}

// ---------------------------------------------------------------------------
// Attention scores + causal softmax -> probs (f16). One wave per 16-row tile.
// Pass A: online row max / sum-exp (WMMA qk, recomputed in pass B).
// Pass B: write probs. P was pre-zeroed so untouched (future) tiles stay 0.
// ---------------------------------------------------------------------------
__global__ void __launch_bounds__(128)
attn_scores_kernel(const f16* __restrict__ q16, const f16* __restrict__ k16,
                   const float* __restrict__ rel_bias, f16* __restrict__ P) {
  int wave = threadIdx.x >> 5;
  int tile = blockIdx.x * 4 + wave;          // 0 .. B*(L/16)-1
  int b  = tile >> 7;                        // L/16 = 128 tiles per batch
  int it = tile & 127;
  int lane = threadIdx.x & 31, nlo = lane & 15, hi = lane >> 4;
  const f16* qb = q16 + (long)b * Lx * Zx;
  const f16* kb = k16 + (long)b * Lx * Zx;
  v16h a0 = load_a(qb, it * 16, 0,  Zx);
  v16h a1 = load_a(qb, it * 16, 32, Zx);
  v16h a2 = load_a(qb, it * 16, 64, Zx);
  v16h a3 = load_a(qb, it * 16, 96, Zx);
  float rm[8], rs[8];
  #pragma unroll
  for (int i = 0; i < 8; ++i) { rm[i] = -1e30f; rs[i] = 0.f; }
  for (int jt = 0; jt <= it; ++jt) {
    v8f c = zero8();
    c = wmma_(a0, load_b(kb, jt * 16, 0,  Zx), c);
    c = wmma_(a1, load_b(kb, jt * 16, 32, Zx), c);
    c = wmma_(a2, load_b(kb, jt * 16, 64, Zx), c);
    c = wmma_(a3, load_b(kb, jt * 16, 96, Zx), c);
    #pragma unroll
    for (int i = 0; i < 8; ++i) {
      int row = it * 16 + i + 8 * hi;
      int col = jt * 16 + nlo;
      float s = (col <= row) ? (c[i] + rel_bias[MAXPOSx - 1 + col - row]) : -1e30f;
      float mx = s;
      mx = fmaxf(mx, __shfl_xor(mx, 1)); mx = fmaxf(mx, __shfl_xor(mx, 2));
      mx = fmaxf(mx, __shfl_xor(mx, 4)); mx = fmaxf(mx, __shfl_xor(mx, 8));
      float nm = fmaxf(rm[i], mx);
      float e  = __expf(s - nm);
      e += __shfl_xor(e, 1); e += __shfl_xor(e, 2);
      e += __shfl_xor(e, 4); e += __shfl_xor(e, 8);
      rs[i] = rs[i] * __expf(rm[i] - nm) + e;
      rm[i] = nm;
    }
  }
  float inv[8];
  #pragma unroll
  for (int i = 0; i < 8; ++i) inv[i] = 1.0f / rs[i];
  f16* Pb = P + (long)b * Lx * Lx;
  for (int jt = 0; jt <= it; ++jt) {
    v8f c = zero8();
    c = wmma_(a0, load_b(kb, jt * 16, 0,  Zx), c);
    c = wmma_(a1, load_b(kb, jt * 16, 32, Zx), c);
    c = wmma_(a2, load_b(kb, jt * 16, 64, Zx), c);
    c = wmma_(a3, load_b(kb, jt * 16, 96, Zx), c);
    #pragma unroll
    for (int i = 0; i < 8; ++i) {
      int row = it * 16 + i + 8 * hi;
      int col = jt * 16 + nlo;
      float s = (col <= row) ? (c[i] + rel_bias[MAXPOSx - 1 + col - row]) : -1e30f;
      Pb[(long)row * Lx + col] = (f16)(__expf(s - rm[i]) * inv[i]);
    }
  }
}

// ---------------------------------------------------------------------------
// GEMM stage 3 (per batch): h = (P @ v) * r  -> hr (B,L,H) f16.
// B-frag source is vT (B,H,L) so K is contiguous per lane.
// ---------------------------------------------------------------------------
__global__ void __launch_bounds__(128)
gemm_h_kernel(const f16* __restrict__ P, const f16* __restrict__ vT,
              const f16* __restrict__ r16, f16* __restrict__ hr) {
  int wave = threadIdx.x >> 5;
  int b = blockIdx.z;
  long m0 = (long)blockIdx.x * 64 + (wave >> 1) * 32;   // rows within L
  long n0 = (long)blockIdx.y * 64 + (wave & 1) * 32;    // H cols
  const f16* A  = P  + (long)b * Lx * Lx;
  const f16* Bp = vT + (long)b * Hx * Lx;
  v8f c00 = zero8(), c01 = zero8(), c10 = zero8(), c11 = zero8();
  for (int k = 0; k < Lx; k += 32) {
    if (k + 32 < Lx) { __builtin_prefetch(Bp + n0 * Lx + k + 32, 0, 1); }
    v16h a0 = load_a(A, m0,      k, Lx);
    v16h a1 = load_a(A, m0 + 16, k, Lx);
    v16h b0 = load_b(Bp, n0,      k, Lx);
    v16h b1 = load_b(Bp, n0 + 16, k, Lx);
    c00 = wmma_(a0, b0, c00); c01 = wmma_(a0, b1, c01);
    c10 = wmma_(a1, b0, c10); c11 = wmma_(a1, b1, c11);
  }
  int lane = threadIdx.x & 31, nlo = lane & 15, hi = lane >> 4;
  v8f cs[4] = {c00, c01, c10, c11};
  int moff[4] = {0, 0, 16, 16}, noff[4] = {0, 16, 0, 16};
  #pragma unroll
  for (int t = 0; t < 4; ++t) {
    #pragma unroll
    for (int i = 0; i < 8; ++i) {
      long row = m0 + moff[t] + i + 8 * hi;
      long col = n0 + noff[t] + nlo;
      long bl  = (long)b * Lx + row;
      hr[bl * Hx + col] = (f16)(cs[t][i] * (float)r16[bl * Hx + col]);
    }
  }
}

// ---------------------------------------------------------------------------
// GEMM stage 4: h2 = silu(hx + hr @ Wh^T); out = x + u*(h2 - x)   (f32 out)
// ---------------------------------------------------------------------------
__global__ void __launch_bounds__(128)
gemm_final_kernel(const f16* __restrict__ A, const f16* __restrict__ W,
                  const f16* __restrict__ hx16, const f16* __restrict__ u16,
                  const float* __restrict__ x, float* __restrict__ out) {
  int wave = threadIdx.x >> 5;
  long m0 = (long)blockIdx.x * 64 + (wave >> 1) * 32;   // token rows [0, B*L)
  long n0 = (long)blockIdx.y * 64 + (wave & 1) * 32;    // D cols
  v8f c00 = zero8(), c01 = zero8(), c10 = zero8(), c11 = zero8();
  for (int k = 0; k < Hx; k += 32) {
    v16h a0 = load_a(A, m0,      k, Hx);
    v16h a1 = load_a(A, m0 + 16, k, Hx);
    v16h b0 = load_b(W, n0,      k, Hx);
    v16h b1 = load_b(W, n0 + 16, k, Hx);
    c00 = wmma_(a0, b0, c00); c01 = wmma_(a0, b1, c01);
    c10 = wmma_(a1, b0, c10); c11 = wmma_(a1, b1, c11);
  }
  int lane = threadIdx.x & 31, nlo = lane & 15, hi = lane >> 4;
  v8f cs[4] = {c00, c01, c10, c11};
  int moff[4] = {0, 0, 16, 16}, noff[4] = {0, 16, 0, 16};
  #pragma unroll
  for (int t = 0; t < 4; ++t) {
    #pragma unroll
    for (int i = 0; i < 8; ++i) {
      long bl  = m0 + moff[t] + i + 8 * hi;
      long col = n0 + noff[t] + nlo;
      float hval = silu_((float)hx16[bl * Dx + col] + cs[t][i]);
      float xv = x[bl * Dx + col];
      float uv = (float)u16[bl * Dx + col];
      out[bl * Dx + col] = xv + uv * (hval - xv);
    }
  }
}

// ---------------------------------------------------------------------------
extern "C" void kernel_launch(void* const* d_in, const int* in_sizes, int n_in,
                              void* d_out, int out_size, void* d_ws, size_t ws_size,
                              hipStream_t stream) {
  (void)in_sizes; (void)n_in; (void)out_size; (void)ws_size;
  const float* x          = (const float*)d_in[0];
  const float* prior_mean = (const float*)d_in[1];
  const float* prior_logv = (const float*)d_in[2];
  const float* tn_weight  = (const float*)d_in[3];
  const float* tn_bias    = (const float*)d_in[4];
  const float* delta      = (const float*)d_in[5];
  const float* alpha      = (const float*)d_in[6];
  const float* ema_beta   = (const float*)d_in[7];
  const float* ema_gamma  = (const float*)d_in[8];
  const float* omega      = (const float*)d_in[9];
  const float* rms_weight = (const float*)d_in[10];
  const float* Wv         = (const float*)d_in[11];
  const float* bv         = (const float*)d_in[12];
  const float* Wmx        = (const float*)d_in[13];
  const float* bmx        = (const float*)d_in[14];
  const float* Wh         = (const float*)d_in[15];
  const float* qk_gamma   = (const float*)d_in[16];
  const float* qk_beta    = (const float*)d_in[17];
  const float* rel_bias   = (const float*)d_in[18];
  float* out = (float*)d_out;

  // Workspace layout (bytes). P16 aliases xn16+mx16 (both dead by then).
  char* ws = (char*)d_ws;
  const size_t SZ_BLD = (size_t)Bx * Lx * Dx * 2;   // 33,554,432
  const size_t SZ_BLH = (size_t)Bx * Lx * Hx * 2;   // 67,108,864
  const size_t SZ_BLZ = (size_t)Bx * Lx * Zx * 2;   //  4,194,304
  size_t off = 0;
  f16* xn16  = (f16*)(ws + off);                 // [0, 32M)
  f16* mx16  = (f16*)(ws + off + SZ_BLD);        // [32M, 64M)
  f16* P16   = (f16*)(ws + off);                 // aliases xn16+mx16 (64M)
  off += 2 * SZ_BLD;
  f16* Wv16  = (f16*)(ws + off); off += (size_t)Hx * Dx * 2;
  f16* Wmx16 = (f16*)(ws + off); off += (size_t)OC * Dx * 2;
  f16* Wh16  = (f16*)(ws + off); off += (size_t)Dx * Hx * 2;
  f16* vT16  = (f16*)(ws + off); off += SZ_BLH;
  f16* u16   = (f16*)(ws + off); off += SZ_BLD;
  f16* r16   = (f16*)(ws + off); off += SZ_BLH;
  f16* hx16  = (f16*)(ws + off); off += SZ_BLD;
  f16* q16   = (f16*)(ws + off); off += SZ_BLZ;
  f16* k16   = (f16*)(ws + off); off += SZ_BLZ;
  f16* hr16  = (f16*)(ws + off); off += SZ_BLH;

  // 0. weight conversion f32 -> f16
  {
    long n;
    n = (long)Hx * Dx;  cvt_f32_f16_kernel<<<(n + 255) / 256, 256, 0, stream>>>(Wv,  Wv16,  n);
    n = (long)OC * Dx;  cvt_f32_f16_kernel<<<(n + 255) / 256, 256, 0, stream>>>(Wmx, Wmx16, n);
    n = (long)Dx * Hx;  cvt_f32_f16_kernel<<<(n + 255) / 256, 256, 0, stream>>>(Wh,  Wh16,  n);
  }
  // 1. timestep norm -> xn16
  tsnorm_kernel<<<Bx * Gx, 32, 0, stream>>>(x, prior_mean, prior_logv, tn_weight, tn_bias, xn16);
  // 2. EMA recurrence -> mx16 (pre-RMS)
  ema_kernel<<<(Bx * Dx) / 256, 256, 0, stream>>>(xn16, delta, alpha, ema_beta, ema_gamma, omega, mx16);
  // 3. RMS norm in place
  rms_kernel<<<(Bx * Lx) / 4, 128, 0, stream>>>(mx16, rms_weight);
  // 4. v projection (silu, transposed output)
  {
    dim3 g(Bx * Lx / 64, Hx / 64);
    gemm_v_kernel<<<g, 128, 0, stream>>>(xn16, Wv16, bv, vT16);
  }
  // 5. base projection with fused gating epilogue
  {
    dim3 g(Bx * Lx / 64, OC / 64);
    gemm_base_kernel<<<g, 128, 0, stream>>>(mx16, Wmx16, bmx, qk_gamma, qk_beta,
                                            u16, q16, k16, r16, hx16);
  }
  // 6. zero attention-prob buffer (causal upper part must be 0 for the GEMM)
  {
    long n16 = ((long)Bx * Lx * Lx * 2) / 16;
    zero16_kernel<<<(n16 + 255) / 256, 256, 0, stream>>>((uint4*)P16, n16);
  }
  // 7. scores + softmax -> P16
  attn_scores_kernel<<<(Bx * (Lx / 16)) / 4, 128, 0, stream>>>(q16, k16, rel_bias, P16);
  // 8. h = (P @ v) * r
  {
    dim3 g(Lx / 64, Hx / 64, Bx);
    gemm_h_kernel<<<g, 128, 0, stream>>>(P16, vT16, r16, hr16);
  }
  // 9. final projection + gated residual -> out
  {
    dim3 g(Bx * Lx / 64, Dx / 64);
    gemm_final_kernel<<<g, 128, 0, stream>>>(hr16, Wh16, hx16, u16, x, out);
  }
}
